// GPTWaterLevelModel_74217034875716
// MI455X (gfx1250) — matrix operand order repaired and checked by reading, not compiled, not run on hardware
//
#include <hip/hip_runtime.h>

// ---- model constants (from reference) ----
#define L_ 8
#define H_ 16
#define E_ 1024
#define T_ 1024
#define B_ 2
#define V_ 800
#define D_ 64
#define M_ (B_ * T_)   // 2048 tokens
#define F_ (4 * E_)    // 4096 MLP width

typedef __attribute__((ext_vector_type(16))) __bf16 v16bf;
typedef __attribute__((ext_vector_type(8)))  float  v8f;
typedef unsigned short u16;
typedef unsigned int   u32;

// POD vectors (HIP's uint4 has ctors -> illegal in unions)
struct __attribute__((aligned(16))) U4 { u32 a, b, c, d; };
struct __attribute__((aligned(8)))  U2 { u32 a, b; };

union BFrag { v16bf v; U4 q[2]; u16 u[16]; };

__device__ __forceinline__ u16 f2bf(float f) {
  union { float f; u32 u; } cv; cv.f = f;
  u32 u = cv.u;
  return (u16)((u + 0x7FFFu + ((u >> 16) & 1u)) >> 16);  // round-to-nearest-even
}

__device__ __forceinline__ v8f wmma_bf16(const BFrag& a, const BFrag& b, v8f c) {
  // (neg_a, A, neg_b, B, c_mod, C, reuse_a, reuse_b)
  return __builtin_amdgcn_wmma_f32_16x16x32_bf16(false, a.v, false, b.v,
                                                 (short)0, c, false, false);
}

// ---- weight prep: W[K,N] f32 -> Wt[N,K] bf16 (32x32 LDS tile transpose) ----
__global__ __launch_bounds__(256) void convert_kernel(
    const float* __restrict__ W, u16* __restrict__ Wt, int K, int N)
{
  __shared__ u16 tile[32][33];  // [n][k], padded against bank conflicts
  const int k0 = blockIdx.x * 32, n0 = blockIdx.y * 32;
  const int c  = threadIdx.x & 31;   // n within tile (coalesced global reads)
  const int r8 = threadIdx.x >> 5;   // k stride-8
#pragma unroll
  for (int i = 0; i < 4; ++i) {
    int k = r8 + i * 8;
    tile[c][k] = f2bf(W[(size_t)(k0 + k) * N + n0 + c]);
  }
  __syncthreads();
  const int rr = threadIdx.x >> 3;        // output row n within tile
  const int cc = (threadIdx.x & 7) * 4;   // 4 consecutive k per thread
  u16* outp = Wt + (size_t)(n0 + rr) * K + k0 + cc;
  U2 w;
  w.a = (u32)tile[rr][cc]     | ((u32)tile[rr][cc + 1] << 16);
  w.b = (u32)tile[rr][cc + 2] | ((u32)tile[rr][cc + 3] << 16);
  *(U2*)outp = w;  // packed 8-byte store, coalesced along k
}

// ---- GEMM: C[M,*] = A(bf16,[M,K]) x Wt(bf16,[N,K]), epilogue by template ----
// Block: 256 threads = 8 waves; each wave computes a 32x64 tile
// (2 A-fragments x 4 B-fragments = 8 WMMAs per 32-wide k-step).
#define MODE_F32_RES   0  // outF[m*N+n] += acc + bias   (residual accumulate)
#define MODE_F32       1  // outF[m*N+n]  = acc + bias   (logits)
#define MODE_BF16_RELU 2  // outH[m*N+n]  = bf16(relu(acc + bias))
#define MODE_BHTD      3  // Q/K scatter -> [B,H,T,D] bf16
#define MODE_BHDT      4  // V  scatter -> [B,H,D,T] bf16 (transposed)

template <int MODE>
__global__ __launch_bounds__(256) void gemm_kernel(
    const u16* __restrict__ A, const u16* __restrict__ Wt,
    const float* __restrict__ bias, float* __restrict__ outF,
    u16* __restrict__ outH, int K, int N)
{
  const int tid  = threadIdx.x;
  const int lane = tid & 31, wv = tid >> 5;
  const int hi   = lane >> 4, l16 = lane & 15;
  const int rowBase = blockIdx.x * 256;   // 8 waves x 32 rows
  const int colBase = blockIdx.y * 64;    // 4 col tiles x 16 cols

  v8f acc[2][4];
  const v8f vz = {0.f, 0.f, 0.f, 0.f, 0.f, 0.f, 0.f, 0.f};
#pragma unroll
  for (int i = 0; i < 2; ++i)
#pragma unroll
    for (int c = 0; c < 4; ++c) acc[i][c] = vz;

  const u16* arow0 = A + (size_t)(rowBase + wv * 32 + l16) * K;
  const u16* arow1 = arow0 + (size_t)16 * K;
  const u16* brow[4];
#pragma unroll
  for (int c = 0; c < 4; ++c) {
    // clamp OOB column tiles (only LM head's last block) to row 0: loads stay
    // valid, results are discarded by the epilogue guard. Keeps the k-loop
    // completely branch-free (no conditional accumulator liveness).
    int ncol = colBase + c * 16;
    if (ncol >= N) ncol = 0;             // wave-uniform, hoisted out of loop
    brow[c] = Wt + (size_t)(ncol + l16) * K + hi * 8;
  }

  for (int k0 = 0; k0 < K; k0 += 32) {
    BFrag a0, a1;
    a0.q[0] = *(const U4*)(arow0 + k0 + hi * 8);
    a0.q[1] = *(const U4*)(arow0 + k0 + 16 + hi * 8);
    a1.q[0] = *(const U4*)(arow1 + k0 + hi * 8);
    a1.q[1] = *(const U4*)(arow1 + k0 + 16 + hi * 8);
#pragma unroll
    for (int c = 0; c < 4; ++c) {
      BFrag b;
      b.q[0] = *(const U4*)(brow[c] + k0);
      b.q[1] = *(const U4*)(brow[c] + k0 + 16);
      acc[0][c] = wmma_bf16(a0, b, acc[0][c]);
      acc[1][c] = wmma_bf16(a1, b, acc[1][c]);
    }
  }

  // epilogue (C/D layout: m = r + 8*hi, n = lane&15)
#pragma unroll
  for (int i = 0; i < 2; ++i) {
#pragma unroll
    for (int c = 0; c < 4; ++c) {
      int nn = colBase + c * 16 + l16;
      if (nn >= N) continue;
      float bv = (MODE == MODE_F32_RES || MODE == MODE_F32 || MODE == MODE_BF16_RELU)
                     ? bias[nn] : 0.f;
#pragma unroll
      for (int r = 0; r < 8; ++r) {
        int mm = rowBase + wv * 32 + i * 16 + r + 8 * hi;
        float val = acc[i][c][r] + bv;
        if (MODE == MODE_F32_RES) {
          float* p = outF + (size_t)mm * N + nn;
          *p = *p + val;
        } else if (MODE == MODE_F32) {
          outF[(size_t)mm * N + nn] = val;
        } else if (MODE == MODE_BF16_RELU) {
          outH[(size_t)mm * N + nn] = f2bf(val > 0.f ? val : 0.f);
        } else if (MODE == MODE_BHTD) {
          int bb = mm >> 10, t = mm & 1023, h = nn >> 6, d = nn & 63;
          outH[(((size_t)(bb * H_ + h)) * T_ + t) * D_ + d] = f2bf(val);
        } else {  // MODE_BHDT
          int bb = mm >> 10, t = mm & 1023, h = nn >> 6, d = nn & 63;
          outH[(((size_t)(bb * H_ + h)) * D_ + d) * T_ + t] = f2bf(val);
        }
      }
    }
  }
}

// ---- causal attention: one block per (16-query tile, b*h) ----
__global__ __launch_bounds__(256) void attn_kernel(
    const u16* __restrict__ Q, const u16* __restrict__ Kmat,
    const u16* __restrict__ Vt, u16* __restrict__ O)
{
  extern __shared__ float smem[];
  float* S   = smem;             // [16][T_] scores -> probabilities
  float* red = smem + 16 * T_;   // [16][16] reduction scratch

  const int tid  = threadIdx.x;
  const int lane = tid & 31, wv = tid >> 5;
  const int hi   = lane >> 4, l16 = lane & 15;
  const int qt   = blockIdx.x;   // query tile
  const int bh   = blockIdx.y;   // b*H + h

  const u16* qb = Q    + (size_t)bh * T_ * D_;
  const u16* kb = Kmat + (size_t)bh * T_ * D_;
  const u16* vb = Vt   + (size_t)bh * D_ * T_;

  // Q A-fragments (16x32, for d0 = 0 and 32)
  BFrag aq[2];
  {
    const u16* qr = qb + (size_t)(qt * 16 + l16) * D_;
#pragma unroll
    for (int g = 0; g < 2; ++g) {
      aq[g].q[0] = *(const U4*)(qr + g * 32 + hi * 8);
      aq[g].q[1] = *(const U4*)(qr + g * 32 + 16 + hi * 8);
    }
  }

  // Phase 1: S = (Q K^T) * D^-0.5 for key tiles j = 0..qt (8 waves round-robin)
  for (int j = wv; j <= qt; j += 8) {
    v8f s = {0.f, 0.f, 0.f, 0.f, 0.f, 0.f, 0.f, 0.f};
#pragma unroll
    for (int g = 0; g < 2; ++g) {
      BFrag b;  // K row-major [T,64] is exactly the "transposed" B layout
      const u16* kr = kb + (size_t)(j * 16 + l16) * D_ + g * 32 + hi * 8;
      b.q[0] = *(const U4*)kr;
      b.q[1] = *(const U4*)(kr + 16);
      s = wmma_bf16(aq[g], b, s);
    }
#pragma unroll
    for (int r = 0; r < 8; ++r)
      S[(r + 8 * hi) * T_ + j * 16 + l16] = s[r] * 0.125f;  // D^-0.5
  }
  __syncthreads();

  // Phase 2: causal softmax (16 threads per row)
  const int row = tid >> 4, sub = tid & 15;
  const int nvalid = qt * 16 + row + 1;
  const int kend = ((qt * 16 + 16) + 31) & ~31;  // 32-multiple cover for phase 3
  float mx = -3.4e38f;
  for (int k = sub; k < nvalid; k += 16) mx = fmaxf(mx, S[row * T_ + k]);
  red[row * 16 + sub] = mx;
  __syncthreads();
  if (sub == 0) {
    float m2 = red[row * 16];
    for (int i = 1; i < 16; ++i) m2 = fmaxf(m2, red[row * 16 + i]);
    red[row * 16] = m2;
  }
  __syncthreads();
  const float rowmax = red[row * 16];
  __syncthreads();
  float sum = 0.f;
  for (int k = sub; k < kend; k += 16) {
    float p = 0.f;
    if (k < nvalid) { p = __expf(S[row * T_ + k] - rowmax); sum += p; }
    S[row * T_ + k] = p;  // masked tail -> exactly 0 (no masking in phase 3)
  }
  red[row * 16 + sub] = sum;
  __syncthreads();
  if (sub == 0) {
    float s2 = 0.f;
    for (int i = 0; i < 16; ++i) s2 += red[row * 16 + i];
    red[row * 16] = 1.f / s2;
  }
  __syncthreads();
  const float rinv = red[row * 16];
  for (int k = sub; k < kend; k += 16) S[row * T_ + k] *= rinv;
  __syncthreads();

  // Phase 3: O = P @ V  (waves 0..3 each own 16 of the 64 d-columns)
  if (wv < 4) {
    v8f o = {0.f, 0.f, 0.f, 0.f, 0.f, 0.f, 0.f, 0.f};
    for (int t0 = 0; t0 < kend; t0 += 32) {
      BFrag a;  // P tile from LDS, f32 -> bf16 on the fly
      const float* sp0 = S + l16 * T_ + t0 + hi * 8;
      const float* sp1 = sp0 + 16;
#pragma unroll
      for (int i = 0; i < 8; ++i) { a.u[i] = f2bf(sp0[i]); a.u[8 + i] = f2bf(sp1[i]); }
      BFrag b;  // V stored transposed [D,T]: keys 8h..8h+7 and 16+8h..16+8h+7 per lane
      const u16* vr = vb + (size_t)(wv * 16 + l16) * T_ + t0 + hi * 8;
      b.q[0] = *(const U4*)vr;
      b.q[1] = *(const U4*)(vr + 16);
      o = wmma_bf16(a, b, o);
    }
    const int bidx = bh >> 4, h = bh & 15;
#pragma unroll
    for (int r = 0; r < 8; ++r) {
      int t = qt * 16 + r + 8 * hi;
      int d = wv * 16 + l16;
      O[((size_t)(bidx * T_ + t)) * E_ + h * D_ + d] = f2bf(o[r]);
    }
  }
}

// ---- embedding: token lookup + lat/lon position GEMV ----
__global__ __launch_bounds__(256) void embed_kernel(
    const float* __restrict__ idx, const float* __restrict__ tok,
    const float* __restrict__ posW, const float* __restrict__ posb,
    float* __restrict__ x)
{
  int i = blockIdx.x;  // token index in [0, B*T)
  float lat = idx[i * 3 + 0], lon = idx[i * 3 + 1], wl = idx[i * 3 + 2];
  int t = (int)roundf(wl * 100.f - 300.f);
  t = t < 0 ? 0 : (t > V_ - 1 ? V_ - 1 : t);
  const float* tr = tok + (size_t)t * E_;
  for (int e = threadIdx.x; e < E_; e += 256)
    x[(size_t)i * E_ + e] = tr[e] + lat * posW[e] + lon * posW[E_ + e] + posb[e];
}

// ---- LayerNorm (population variance), emits bf16 activations ----
__global__ __launch_bounds__(256) void ln_kernel(
    const float* __restrict__ x, const float* __restrict__ g,
    const float* __restrict__ b, u16* __restrict__ out)
{
  __shared__ float red[256];
  __shared__ float s_mean, s_rstd;
  int row = blockIdx.x;
  const float* xr = x + (size_t)row * E_;
  float s = 0.f, s2 = 0.f;
  for (int e = threadIdx.x; e < E_; e += 256) { float v = xr[e]; s += v; s2 += v * v; }
  red[threadIdx.x] = s; __syncthreads();
  for (int o = 128; o > 0; o >>= 1) {
    if (threadIdx.x < o) red[threadIdx.x] += red[threadIdx.x + o];
    __syncthreads();
  }
  if (threadIdx.x == 0) s_mean = red[0] * (1.f / E_);
  __syncthreads();
  red[threadIdx.x] = s2; __syncthreads();
  for (int o = 128; o > 0; o >>= 1) {
    if (threadIdx.x < o) red[threadIdx.x] += red[threadIdx.x + o];
    __syncthreads();
  }
  if (threadIdx.x == 0) {
    float var = red[0] * (1.f / E_) - s_mean * s_mean;
    s_rstd = rsqrtf(var + 1e-5f);
  }
  __syncthreads();
  float mean = s_mean, rstd = s_rstd;
  for (int e = threadIdx.x; e < E_; e += 256)
    out[(size_t)row * E_ + e] = f2bf((xr[e] - mean) * rstd * g[e] + b[e]);
}

extern "C" void kernel_launch(void* const* d_in, const int* in_sizes, int n_in,
                              void* d_out, int out_size, void* d_ws, size_t ws_size,
                              hipStream_t stream) {
  (void)in_sizes; (void)n_in; (void)out_size; (void)ws_size;
  const float* idx  = (const float*)d_in[0];
  const float* tok  = (const float*)d_in[1];
  const float* posW = (const float*)d_in[2];
  const float* posb = (const float*)d_in[3];
  const float* Wq   = (const float*)d_in[4];
  const float* Wk   = (const float*)d_in[5];
  const float* Wvv  = (const float*)d_in[6];
  const float* Wo   = (const float*)d_in[7];
  const float* bo   = (const float*)d_in[8];
  const float* W1   = (const float*)d_in[9];
  const float* b1   = (const float*)d_in[10];
  const float* W2   = (const float*)d_in[11];
  const float* b2   = (const float*)d_in[12];
  const float* ln1g = (const float*)d_in[13];
  const float* ln1b = (const float*)d_in[14];
  const float* ln2g = (const float*)d_in[15];
  const float* ln2b = (const float*)d_in[16];
  const float* lnfg = (const float*)d_in[17];
  const float* lnfb = (const float*)d_in[18];
  const float* lmW  = (const float*)d_in[19];
  const float* lmb  = (const float*)d_in[20];
  float* out = (float*)d_out;

  char* ws = (char*)d_ws;
  size_t off = 0;
  auto alloc = [&](size_t bytes) {
    void* p = ws + off;
    off += (bytes + 255) & ~(size_t)255;
    return p;
  };
  // activations (~44 MB)
  float* xbuf  = (float*)alloc((size_t)M_ * E_ * 4);  // fp32 residual stream
  u16* hbuf    = (u16*)  alloc((size_t)M_ * E_ * 2);  // bf16 LN output
  u16* qbuf    = (u16*)  alloc((size_t)M_ * E_ * 2);  // [B,H,T,D]
  u16* kbuf    = (u16*)  alloc((size_t)M_ * E_ * 2);  // [B,H,T,D]
  u16* vbuf    = (u16*)  alloc((size_t)M_ * E_ * 2);  // [B,H,D,T]
  u16* abuf    = (u16*)  alloc((size_t)M_ * E_ * 2);  // attention out [B,T,E]
  u16* midbuf  = (u16*)  alloc((size_t)M_ * F_ * 2);  // MLP mid [M,4E]
  // per-layer bf16 transposed weight scratch (~25 MB, reused each layer)
  u16* wtq = (u16*)alloc((size_t)E_ * E_ * 2);
  u16* wtk = (u16*)alloc((size_t)E_ * E_ * 2);
  u16* wtv = (u16*)alloc((size_t)E_ * E_ * 2);
  u16* wto = (u16*)alloc((size_t)E_ * E_ * 2);
  u16* wt1 = (u16*)alloc((size_t)F_ * E_ * 2);  // W1^T [4E, E]
  u16* wt2 = (u16*)alloc((size_t)E_ * F_ * 2);  // W2^T [E, 4E]

  dim3 blk(256);
  embed_kernel<<<dim3(M_), blk, 0, stream>>>(idx, tok, posW, posb, xbuf);

  const size_t attn_smem = (size_t)(16 * T_ + 16 * 16) * sizeof(float);
  const dim3 gEE(E_ / 32, E_ / 32);   // convert grids
  const dim3 gEF(E_ / 32, F_ / 32);
  const dim3 gFE(F_ / 32, E_ / 32);

  for (int l = 0; l < L_; ++l) {
    // convert this layer's weights: f32 [K,N] -> bf16 [N,K] (serialized on stream)
    convert_kernel<<<gEE, blk, 0, stream>>>(Wq  + (size_t)l * E_ * E_, wtq, E_, E_);
    convert_kernel<<<gEE, blk, 0, stream>>>(Wk  + (size_t)l * E_ * E_, wtk, E_, E_);
    convert_kernel<<<gEE, blk, 0, stream>>>(Wvv + (size_t)l * E_ * E_, wtv, E_, E_);
    convert_kernel<<<gEE, blk, 0, stream>>>(Wo  + (size_t)l * E_ * E_, wto, E_, E_);
    convert_kernel<<<gEF, blk, 0, stream>>>(W1  + (size_t)l * E_ * F_, wt1, E_, F_);
    convert_kernel<<<gFE, blk, 0, stream>>>(W2  + (size_t)l * F_ * E_, wt2, F_, E_);

    ln_kernel<<<dim3(M_), blk, 0, stream>>>(xbuf, ln1g + l * E_, ln1b + l * E_, hbuf);
    gemm_kernel<MODE_BHTD><<<dim3(M_/256, E_/64), blk, 0, stream>>>(hbuf, wtq, nullptr, nullptr, qbuf, E_, E_);
    gemm_kernel<MODE_BHTD><<<dim3(M_/256, E_/64), blk, 0, stream>>>(hbuf, wtk, nullptr, nullptr, kbuf, E_, E_);
    gemm_kernel<MODE_BHDT><<<dim3(M_/256, E_/64), blk, 0, stream>>>(hbuf, wtv, nullptr, nullptr, vbuf, E_, E_);
    attn_kernel<<<dim3(T_/16, B_*H_), blk, attn_smem, stream>>>(qbuf, kbuf, vbuf, abuf);
    gemm_kernel<MODE_F32_RES><<<dim3(M_/256, E_/64), blk, 0, stream>>>(abuf, wto, bo + (size_t)l * E_, xbuf, nullptr, E_, E_);
    ln_kernel<<<dim3(M_), blk, 0, stream>>>(xbuf, ln2g + l * E_, ln2b + l * E_, hbuf);
    gemm_kernel<MODE_BF16_RELU><<<dim3(M_/256, F_/64), blk, 0, stream>>>(hbuf, wt1, b1 + (size_t)l * F_, nullptr, midbuf, E_, F_);
    gemm_kernel<MODE_F32_RES><<<dim3(M_/256, E_/64), blk, 0, stream>>>(midbuf, wt2, b2 + (size_t)l * E_, xbuf, nullptr, F_, E_);
  }

  ln_kernel<<<dim3(M_), blk, 0, stream>>>(xbuf, lnfg, lnfb, hbuf);
  // LM head: reuse wtq scratch (800*1024 bf16 fits in E*E slot); V=800 = 25*32
  convert_kernel<<<dim3(E_ / 32, V_ / 32), blk, 0, stream>>>(lmW, wtq, E_, V_);
  gemm_kernel<MODE_F32><<<dim3(M_/256, (V_ + 63)/64), blk, 0, stream>>>(hbuf, wtq, lmb, out, nullptr, E_, V_);
}
